// NodeNNConvModel_55825984913731
// MI455X (gfx1250) — compile-verified, matrix operand order checked by reading, not run on hardware
//
#include <hip/hip_runtime.h>
#include <math.h>

typedef __attribute__((ext_vector_type(16))) _Float16 v16h;
typedef __attribute__((ext_vector_type(8)))  float    v8f;
typedef __attribute__((ext_vector_type(2)))  _Float16 h2;

#define NNODES 50000
#define NEDGES 200000

// slope in (0,1): leaky(x) = max(x, s*x)  (branchless, no cndmask chain)
__device__ __forceinline__ float leakyf(float x, float s) { return fmaxf(x, s * x); }

// ---------------- utility kernels ----------------
__global__ void zero_kernel(float* p, long n) {
  long i = (long)blockIdx.x * blockDim.x + threadIdx.x;
  long st = (long)gridDim.x * blockDim.x;
  for (; i < n; i += st) p[i] = 0.0f;
}

__global__ void colstats_kernel(const float* __restrict__ src, int R, int C,
                                float* __restrict__ sum, float* __restrict__ sq) {
  long tid = (long)blockIdx.x * blockDim.x + threadIdx.x;
  long total = (long)gridDim.x * blockDim.x;
  long nslots = total / C;
  int c = (int)(tid % C);
  long slot = tid / C;
  if (slot >= nslots) return;
  float s = 0.f, q = 0.f;
  for (long r = slot; r < R; r += nslots) {
    float v = src[r * C + c];
    s += v; q += v * v;
  }
  atomicAdd(&sum[c], s);
  atomicAdd(&sq[c], q);
}

__global__ void bn_finalize_kernel(const float* __restrict__ sum, const float* __restrict__ sq,
                                   const float* __restrict__ g, const float* __restrict__ b,
                                   float invR, int C, float* __restrict__ scale,
                                   float* __restrict__ shift) {
  int c = blockIdx.x * blockDim.x + threadIdx.x;
  if (c >= C) return;
  float m = sum[c] * invR;
  float var = sq[c] * invR - m * m;
  float sc = g[c] * rsqrtf(var + 1e-5f);
  scale[c] = sc;
  shift[c] = b[c] - m * sc;
}

__global__ void bn_apply_node_kernel(const float* __restrict__ x, const float* __restrict__ scale,
                                     const float* __restrict__ shift, float* __restrict__ h) {
  int i = blockIdx.x * blockDim.x + threadIdx.x;
  if (i >= NNODES * 16) return;
  int n = i >> 4, k = i & 15;
  h[n * 32 + k] = x[i] * scale[k] + shift[k];   // h buffers use a fixed stride of 32
}

__global__ void bn_apply_edge_kernel(const float* __restrict__ e, const float* __restrict__ scale,
                                     const float* __restrict__ shift, float* __restrict__ en,
                                     long total) {
  long i = (long)blockIdx.x * blockDim.x + threadIdx.x;
  long st = (long)gridDim.x * blockDim.x;
  for (; i < total; i += st) {
    int k = (int)(i % 19);
    en[i] = e[i] * scale[k] + shift[k];
  }
}

// Pre-swizzle an f32 (Krows x Ncols) weight into f16 WMMA B-fragment order.
// Fragment layout (ISA 7.12.2, 16-bit B 32x16): tile t, lane = (n%16) + 16*((k>>3)&1),
// half-slot h = (k&7) + 8*(k>=16).  Out index = (t*32+lane)*16 + h.  K padded to 32 with 0.
__global__ void swizzle_kernel(const float* __restrict__ w, int Krows, int Ncols,
                               _Float16* __restrict__ out, int total) {
  int idx = blockIdx.x * blockDim.x + threadIdx.x;
  if (idx >= total) return;
  int hh = idx & 15;
  int lane = (idx >> 4) & 31;
  int t = idx >> 9;
  int n = t * 16 + (lane & 15);
  int k = (hh & 7) + 8 * ((lane >> 4) & 1) + ((hh >> 3) << 4);
  float v = (k < Krows && n < Ncols) ? w[(size_t)k * Ncols + n] : 0.0f;
  out[idx] = (_Float16)v;
}

// h_next = h @ root_w + conv_b  (edge-conv kernel then atomically adds the aggregated messages)
__global__ void root_init_kernel(const float* __restrict__ h, const float* __restrict__ rw,
                                 const float* __restrict__ cb, float* __restrict__ hout, int nin) {
  int i = blockIdx.x * blockDim.x + threadIdx.x;
  if (i >= NNODES * 32) return;
  int n = i >> 5, o = i & 31;
  float acc = cb[o];
  for (int k = 0; k < nin; ++k) acc += h[n * 32 + k] * rw[k * 32 + o];
  hout[i] = acc;
}

// ---------------- fused edge-conditioned convolution (WMMA) ----------------
// Per 16-edge group (one wave):
//   a  = leaky(e_n @ w1 + b1)               (WMMA #1, K padded 19->32)
//   we = leaky(a @ w2 + b2)  tile-by-tile   (WMMA #2, 2*NIN tiles of 16 cols)
//   msg[e,o] = sum_i h[src[e],i] * we[e, i*32+o]   fused into the tile epilogue
//   atomicAdd(hout[dst[e]], msg[e,:])
// Weight fragments + biases are staged once per block into LDS (MI455X: 320KB/WGP).
template <int NIN>
__global__ void edge_conv_kernel(const float* __restrict__ en,    // E x 19 normalized
                                 const float* __restrict__ h,     // N x 32
                                 const int* __restrict__ eidx,    // 2 x E
                                 const _Float16* __restrict__ w1f,// (NIN/16)*512 halves
                                 const float* __restrict__ b1,    // NIN
                                 const _Float16* __restrict__ w2f,// (2*NIN)*512 halves
                                 const float* __restrict__ b2,    // NIN*32
                                 float* __restrict__ hout,        // N x 32 (pre-init root)
                                 int nedges) {
  constexpr int NT = 2 * NIN;
  constexpr int T1 = NIN / 16;
  __shared__ _Float16 w2s[NT * 512];     // 64KB (NIN=32) / 32KB (NIN=16)
  __shared__ _Float16 w1s[T1 * 512];
  __shared__ float b2s[NIN * 32];
  __shared__ float b1s[NIN];
  __shared__ _Float16 a_sh[8][16 * 32];  // wave-private a tile (f16)
  __shared__ float hs_sh[8][16 * 32];    // wave-private h[src] tile

  // ---- stage weights into LDS once per block ----
  for (int i = threadIdx.x; i < NT * 512; i += blockDim.x) w2s[i] = w2f[i];
  for (int i = threadIdx.x; i < T1 * 512; i += blockDim.x) w1s[i] = w1f[i];
  for (int i = threadIdx.x; i < NIN * 32; i += blockDim.x) b2s[i] = b2[i];
  if (threadIdx.x < NIN) b1s[threadIdx.x] = b1[threadIdx.x];
  __syncthreads();

  const int wv = threadIdx.x >> 5;
  const int lane = threadIdx.x & 31;
  const int sel = lane >> 4;     // lane half: which K / M sub-range this lane owns
  const int m = lane & 15;       // edge row within group / column within tile
  const int ngroups = nedges >> 4;
  const int gstride = (gridDim.x * blockDim.x) >> 5;

  for (int g = (blockIdx.x * blockDim.x + threadIdx.x) >> 5; g < ngroups; g += gstride) {
    const int ebase = g << 4;
    const int myedge = ebase + m;
    const int snode = eidx[myedge];

    // h[src] tile -> LDS (16 edges x NIN feats, stride 32)
    if (sel * 16 < NIN) {
#pragma unroll
      for (int jj = 0; jj < 16; ++jj) {
        int k = sel * 16 + jj;
        hs_sh[wv][m * 32 + k] = h[(size_t)snode * 32 + k];
      }
    }

    // A-fragment of e_n (16x32, K 19..31 zero); clamped unconditional load, no exec branches
    v16h ef;
#pragma unroll
    for (int hh = 0; hh < 16; ++hh) {
      int k = (hh & 7) + 8 * sel + ((hh >> 3) << 4);
      int kc = (k < 19) ? k : 0;
      float fv = en[(size_t)myedge * 19 + kc];
      fv = (k < 19) ? fv : 0.0f;
      ef[hh] = (_Float16)fv;
    }

    // GEMM #1: a = leaky(e_n @ w1 + b1) -> f16 into wave-private LDS
#pragma unroll
    for (int t1 = 0; t1 < T1; ++t1) {
      v16h bf;
      const h2* bp = (const h2*)(w1s + (t1 * 32 + lane) * 16);
#pragma unroll
      for (int p = 0; p < 8; ++p) { h2 pr = bp[p]; bf[2 * p] = pr.x; bf[2 * p + 1] = pr.y; }
      v8f c = {};
      v8f d = __builtin_amdgcn_wmma_f32_16x16x32_f16(false, ef, false, bf, (short)0, c, false, false);
      float b1v = b1s[t1 * 16 + m];
#pragma unroll
      for (int vv = 0; vv < 8; ++vv) {
        float a = leakyf(d[vv] + b1v, 0.1f);
        int m2 = vv + 8 * sel;
        a_sh[wv][m2 * 32 + t1 * 16 + m] = (_Float16)a;   // row m2, feature column
      }
    }
    // LDS ops from the same wave complete in order -> safe to read a_sh back.

    // A-fragment of a (clamped address + select 0 so poisoned LDS never feeds the WMMA)
    v16h af;
#pragma unroll
    for (int p = 0; p < 8; ++p) {
      int bk = 8 * sel + ((p >> 2) << 4) + 2 * (p & 3);
      h2 pr = *(const h2*)&a_sh[wv][m * 32 + (bk & (NIN - 1))];
      if (bk >= NIN) pr = (h2){};
      af[2 * p] = pr.x;
      af[2 * p + 1] = pr.y;
    }

    // GEMM #2: iterate input features i; tile pair (2i, 2i+1) shares the h_src factors
    v8f macc0 = {};  // outputs o = 0..15
    v8f macc1 = {};  // outputs o = 16..31
#pragma unroll 2
    for (int i = 0; i < NIN; ++i) {
      float hf[8];
#pragma unroll
      for (int vv = 0; vv < 8; ++vv) hf[vv] = hs_sh[wv][(vv + 8 * sel) * 32 + i];
#pragma unroll
      for (int half = 0; half < 2; ++half) {
        int t = 2 * i + half;
        v16h bf;
        const h2* bp = (const h2*)(w2s + (t * 32 + lane) * 16);
#pragma unroll
        for (int p = 0; p < 8; ++p) { h2 pr = bp[p]; bf[2 * p] = pr.x; bf[2 * p + 1] = pr.y; }
        v8f c = {};
        v8f d = __builtin_amdgcn_wmma_f32_16x16x32_f16(false, af, false, bf, (short)0, c, false, false);
        float b2v = b2s[t * 16 + m];
#pragma unroll
        for (int vv = 0; vv < 8; ++vv) {
          float w = leakyf(d[vv] + b2v, 0.1f);
          if (half) macc1[vv] += w * hf[vv];
          else      macc0[vv] += w * hf[vv];
        }
      }
    }

    // scatter messages: agg[dst] += msg
#pragma unroll
    for (int vv = 0; vv < 8; ++vv) {
      int m2 = vv + 8 * sel;
      int dnode = eidx[nedges + ebase + m2];
      atomicAdd(&hout[(size_t)dnode * 32 + m], macc0[vv]);
      atomicAdd(&hout[(size_t)dnode * 32 + 16 + m], macc1[vv]);
    }
  }
}

// ---------------- MetaLayer edge model + node-model scatter ----------------
__global__ void edge_model_kernel(const float* __restrict__ h, const float* __restrict__ en,
                                  const int* __restrict__ eidx,
                                  const float* __restrict__ em_w1, const float* __restrict__ em_b1,
                                  const float* __restrict__ em_w2, const float* __restrict__ em_b2,
                                  float* __restrict__ s, float* __restrict__ cnt, int nedges) {
  __shared__ float w1s[83 * 16];
  __shared__ float b1s[16], w2s[32], b2s[2];
  for (int i = threadIdx.x; i < 83 * 16; i += blockDim.x) w1s[i] = em_w1[i];
  if (threadIdx.x < 16) b1s[threadIdx.x] = em_b1[threadIdx.x];
  if (threadIdx.x < 32) w2s[threadIdx.x] = em_w2[threadIdx.x];
  if (threadIdx.x < 2) b2s[threadIdx.x] = em_b2[threadIdx.x];
  __syncthreads();

  int e = blockIdx.x * blockDim.x + threadIdx.x;
  int st = gridDim.x * blockDim.x;
  for (; e < nedges; e += st) {
    int sn = eidx[e], dn = eidx[nedges + e];
    float t[16];
#pragma unroll
    for (int j = 0; j < 16; ++j) t[j] = b1s[j];
    for (int q = 0; q < 32; ++q) {
      float fq = h[(size_t)sn * 32 + q];
#pragma unroll
      for (int j = 0; j < 16; ++j) t[j] += fq * w1s[q * 16 + j];
    }
    float hd[32];
    for (int q = 0; q < 32; ++q) {
      hd[q] = h[(size_t)dn * 32 + q];
#pragma unroll
      for (int j = 0; j < 16; ++j) t[j] += hd[q] * w1s[(32 + q) * 16 + j];
    }
    for (int q = 0; q < 19; ++q) {
      float fq = en[(size_t)e * 19 + q];
#pragma unroll
      for (int j = 0; j < 16; ++j) t[j] += fq * w1s[(64 + q) * 16 + j];
    }
    float e0 = b2s[0], e1 = b2s[1];
#pragma unroll
    for (int j = 0; j < 16; ++j) {
      float tj = leakyf(t[j], 0.1f);
      e0 += tj * w2s[2 * j];
      e1 += tj * w2s[2 * j + 1];
    }
    // scatter nm_in = [h[dst], ep] by src, plus edge count
    for (int q = 0; q < 32; ++q) atomicAdd(&s[(size_t)sn * 34 + q], hd[q]);
    atomicAdd(&s[(size_t)sn * 34 + 32], e0);
    atomicAdd(&s[(size_t)sn * 34 + 33], e1);
    atomicAdd(&cnt[sn], 1.0f);
  }
}

__global__ void node_model_kernel(const float* __restrict__ s, const float* __restrict__ cnt,
                                  const float* __restrict__ w1, const float* __restrict__ b1,
                                  const float* __restrict__ w2, const float* __restrict__ b2,
                                  float* __restrict__ out) {
  int n = blockIdx.x * blockDim.x + threadIdx.x;
  if (n >= NNODES) return;
  float inv = 1.0f / fmaxf(cnt[n], 1.0f);
  float t[16];
#pragma unroll
  for (int j = 0; j < 16; ++j) t[j] = b1[j];
  for (int q = 0; q < 34; ++q) {
    float mq = s[(size_t)n * 34 + q] * inv;
#pragma unroll
    for (int j = 0; j < 16; ++j) t[j] += mq * w1[q * 16 + j];
  }
  float o0 = b2[0], o1 = b2[1];
#pragma unroll
  for (int j = 0; j < 16; ++j) {
    float tj = leakyf(t[j], 0.12f);
    o0 += tj * w2[2 * j];
    o1 += tj * w2[2 * j + 1];
  }
  float mx = fmaxf(o0, o1);
  float lse = mx + logf(expf(o0 - mx) + expf(o1 - mx));
  out[2 * n] = o0 - lse;
  out[2 * n + 1] = o1 - lse;
}

// ---------------- host side ----------------
static inline int cdiv(long a, int b) { return (int)((a + b - 1) / b); }

extern "C" void kernel_launch(void* const* d_in, const int* in_sizes, int n_in,
                              void* d_out, int out_size, void* d_ws, size_t ws_size,
                              hipStream_t stream) {
  (void)in_sizes; (void)n_in; (void)out_size; (void)ws_size;
  constexpr int N = NNODES, E = NEDGES;

  const float* x     = (const float*)d_in[0];
  const int*   eidx  = (const int*)d_in[1];
  const float* e     = (const float*)d_in[2];
  const float* bn_ng = (const float*)d_in[4];
  const float* bn_nb = (const float*)d_in[5];
  const float* bn_eg = (const float*)d_in[6];
  const float* bn_eb = (const float*)d_in[7];
  const float* nn_w1[3] = {(const float*)d_in[8],  (const float*)d_in[14], (const float*)d_in[20]};
  const float* nn_b1[3] = {(const float*)d_in[9],  (const float*)d_in[15], (const float*)d_in[21]};
  const float* nn_w2[3] = {(const float*)d_in[10], (const float*)d_in[16], (const float*)d_in[22]};
  const float* nn_b2[3] = {(const float*)d_in[11], (const float*)d_in[17], (const float*)d_in[23]};
  const float* root_w[3]= {(const float*)d_in[12], (const float*)d_in[18], (const float*)d_in[24]};
  const float* conv_b[3]= {(const float*)d_in[13], (const float*)d_in[19], (const float*)d_in[25]};
  const float* em_w1 = (const float*)d_in[26];
  const float* em_b1 = (const float*)d_in[27];
  const float* em_w2 = (const float*)d_in[28];
  const float* em_b2 = (const float*)d_in[29];
  const float* nm_w1 = (const float*)d_in[30];
  const float* nm_b1 = (const float*)d_in[31];
  const float* nm_w2 = (const float*)d_in[32];
  const float* nm_b2 = (const float*)d_in[33];
  float* out = (float*)d_out;

  // workspace layout (float offsets)
  float* ws = (float*)d_ws;
  const long OFF_STATS = 0;                         // 160 floats
  const long OFF_EN    = 160;                       // E*19
  const long OFF_HA    = OFF_EN + (long)E * 19;     // N*32
  const long OFF_HB    = OFF_HA + (long)N * 32;     // N*32
  const long OFF_W1F   = OFF_HB + (long)N * 32;     // 1024 halves (512 floats)
  const long OFF_W2F   = OFF_W1F + 512;             // 32768 halves (16384 floats)
  const long OFF_S     = OFF_W2F + 16384;           // N*34
  const long OFF_CNT   = OFF_S + (long)N * 34;      // N
  float* st   = ws + OFF_STATS;
  float* en   = ws + OFF_EN;
  float* hA   = ws + OFF_HA;
  float* hB   = ws + OFF_HB;
  _Float16* w1f = (_Float16*)(ws + OFF_W1F);
  _Float16* w2f = (_Float16*)(ws + OFF_W2F);
  float* sbuf = ws + OFF_S;
  float* cbuf = ws + OFF_CNT;
  // stats region: node sum@0 sq@16 scale@32 shift@48 | edge sum@64 sq@83 scale@102 shift@121

  // 1) BatchNorm (training mode, biased variance)
  zero_kernel<<<1, 256, 0, stream>>>(st, 160);
  colstats_kernel<<<128, 256, 0, stream>>>(x, N, 16, st + 0, st + 16);
  colstats_kernel<<<256, 256, 0, stream>>>(e, E, 19, st + 64, st + 83);
  bn_finalize_kernel<<<1, 64, 0, stream>>>(st + 0, st + 16, bn_ng, bn_nb, 1.0f / N, 16, st + 32, st + 48);
  bn_finalize_kernel<<<1, 64, 0, stream>>>(st + 64, st + 83, bn_eg, bn_eb, 1.0f / E, 19, st + 102, st + 121);
  bn_apply_node_kernel<<<cdiv((long)N * 16, 256), 256, 0, stream>>>(x, st + 32, st + 48, hA);
  bn_apply_edge_kernel<<<512, 256, 0, stream>>>(e, st + 102, st + 121, en, (long)E * 19);

  // 2) three NNConv layers (ping-pong hA <-> hB)
  const float* hin = hA;
  float* hpair[2] = {hB, hA};
  for (int l = 0; l < 3; ++l) {
    int nin = (l == 0) ? 16 : 32;
    int T1 = nin / 16, NT = 2 * nin;
    float* hout = hpair[l & 1];
    swizzle_kernel<<<cdiv(T1 * 512, 256), 256, 0, stream>>>(nn_w1[l], 19, nin, w1f, T1 * 512);
    swizzle_kernel<<<cdiv(NT * 512, 256), 256, 0, stream>>>(nn_w2[l], nin, nin * 32, w2f, NT * 512);
    root_init_kernel<<<cdiv((long)N * 32, 256), 256, 0, stream>>>(hin, root_w[l], conv_b[l], hout, nin);
    if (nin == 16)
      edge_conv_kernel<16><<<640, 256, 0, stream>>>(en, hin, eidx, w1f, nn_b1[l], w2f, nn_b2[l], hout, E);
    else
      edge_conv_kernel<32><<<640, 256, 0, stream>>>(en, hin, eidx, w1f, nn_b1[l], w2f, nn_b2[l], hout, E);
    hin = hout;
  }
  // after 3 layers hin == hB

  // 3) MetaLayer: edge model + scatter-mean inputs
  zero_kernel<<<512, 256, 0, stream>>>(sbuf, (long)N * 34 + N);  // s and cnt are contiguous
  edge_model_kernel<<<512, 256, 0, stream>>>(hin, en, eidx, em_w1, em_b1, em_w2, em_b2, sbuf, cbuf, E);

  // 4) node model + log_softmax -> d_out
  node_model_kernel<<<cdiv(N, 256), 256, 0, stream>>>(sbuf, cbuf, nm_w1, nm_b1, nm_w2, nm_b2, out);
}
